// MyVit_68186900791590
// MI455X (gfx1250) — compile-verified, matrix operand order checked by reading, not run on hardware
//
#include <hip/hip_runtime.h>
#include <hip/hip_bf16.h>
#include <math.h>

// ---------------- model constants ----------------
#define B_      64
#define S_      197
#define D_      768
#define NH_     12
#define DH_     64
#define FF_     3072
#define L_      12
#define P_      14
#define NPATCH  196          // 14*14
#define SP_M    208          // S padded to multiple of 16 (attn row dim)
#define SP_K    256          // S padded to multiple of 64 (attn K dim; K%64==0 for ping-pong loop)

typedef __attribute__((ext_vector_type(16))) __bf16 bf16x16;
typedef __attribute__((ext_vector_type(8)))  float  f32x8;
typedef __attribute__((ext_vector_type(4)))  unsigned int u32x4;
// explicitly global (address_space(1)) vector pointee: forces global_load_b128
typedef __attribute__((address_space(1))) const u32x4 gc_u32x4;

// ---------------------------------------------------------------------------
// Raw fragment load: two 16-byte chunks at p and p+16 halves, via explicit
// global address space (avoids flat_load + DScnt coupling when the addrspace
// inference pass loses track of array-held pointers).
// With per-lane base pointers set up per the CDNA5 16x16x32 bf16 layout
// (lane<16: row = base+lane, K-halves {0..7,16..23}; lane>=16: +8 K offset),
// this yields a valid A fragment (or B fragment for W[N,K] row-major).
// ---------------------------------------------------------------------------
__device__ __forceinline__ bf16x16 frag_ld(const __bf16* p) {
  const gc_u32x4* gp = (const gc_u32x4*)(unsigned long long)p;
  union { bf16x16 v; u32x4 u[2]; } f;
  f.u[0] = gp[0];
  f.u[1] = gp[2];   // +32 bytes == +16 halves
  return f.v;
}

// per-lane fragment base pointer for row-major [rows, ld], clamped row
__device__ __forceinline__ const __bf16* frag_base(const __bf16* base, int row0, int ld, int maxrow) {
  const int lane = threadIdx.x & 31;
  int r = row0 + (lane & 15);
  if (r > maxrow) r = maxrow;
  return base + (size_t)r * (size_t)ld + ((lane >> 4) << 3);
}

// ---------------------------------------------------------------------------
// Generic batched bf16 WMMA GEMM:  C = act(A @ W.T + bias [+ resid])
//   A: [M,K] bf16 row-major (lda); B: W[N,K] bf16 row-major (ldb)
//   REQUIRES K % 64 == 0.  Buffers must have >= 256B tail slack (the ping-pong
//   loop's final prefetch over-reads up to 62B past the last row).
//   BN: B-fragments per wave (2 -> wave 32x32, block 128x64;
//                             4 -> wave 32x64, block 128x128)
//   CMODE=1: remap C row m -> m + m/196 + 1 (ViT token insert, skip cls rows)
//   CT=1:    transposed store for V: C[(b*NH+h)*DH + col][s] with ld SP_K
// 256 threads = 8 waves: waveM = wid>>1 (4 x 32 rows), waveN = wid&1.
// K-loop: explicit two-stage register ping-pong; next-trip fragments land
// directly in the stage-A buffers (no rotation copies, no WMMA->VALU nops).
// ---------------------------------------------------------------------------
template<int BN, int OUT_BF16, int GELU_ACT, int RESID, int CMODE, int CT>
__launch_bounds__(256, 1)
__global__ void gemm_wmma_kernel(const __bf16* __restrict__ A,  int lda, long long sA1, long long sA2,
                                 const __bf16* __restrict__ Bm, int ldb, long long sB1, long long sB2,
                                 void* __restrict__ Cv,         int ldc, long long sC1, long long sC2,
                                 const float* __restrict__ bias, long long sBi1, long long sBi2,
                                 const float* __restrict__ resid,
                                 int M, int N, int K, int batch2)
{
  const int z  = blockIdx.z;
  const int z1 = z / batch2;
  const int z2 = z - z1 * batch2;
  A  += (size_t)z1 * sA1 + (size_t)z2 * sA2;
  Bm += (size_t)z1 * sB1 + (size_t)z2 * sB2;
  const size_t coff = (size_t)z1 * sC1 + (size_t)z2 * sC2;
  const float* biasp = bias ? (bias + (size_t)z1 * sBi1 + (size_t)z2 * sBi2) : nullptr;

  const int wid  = threadIdx.x >> 5;
  const int lane = threadIdx.x & 31;
  const int m0 = blockIdx.y * 128 + (wid >> 1) * 32;
  const int n0 = blockIdx.x * (BN * 32) + (wid & 1) * (BN * 16);

  // hoisted per-lane fragment base pointers (row-clamped)
  const __bf16* pa0 = frag_base(A, m0,      lda, M - 1);
  const __bf16* pa1 = frag_base(A, m0 + 16, lda, M - 1);
  const __bf16* pb[BN];
#pragma unroll
  for (int j = 0; j < BN; ++j) pb[j] = frag_base(Bm, n0 + j * 16, ldb, N - 1);

  f32x8 acc[2][BN] = {};

  // ---- stage-A buffers: fragments for k = 0 ----
  bf16x16 a0c = frag_ld(pa0);
  bf16x16 a1c = frag_ld(pa1);
  bf16x16 bcur[BN];
#pragma unroll
  for (int j = 0; j < BN; ++j) bcur[j] = frag_ld(pb[j]);

  // ---- two-stage ping-pong main loop (K % 64 == 0) ----
  for (int k0 = 0; k0 < K; k0 += 64) {
    // prefetch stage B (k0+32)
    bf16x16 a0n = frag_ld(pa0 + 32);
    bf16x16 a1n = frag_ld(pa1 + 32);
    bf16x16 bnx[BN];
#pragma unroll
    for (int j = 0; j < BN; ++j) bnx[j] = frag_ld(pb[j] + 32);
    // compute stage A
#pragma unroll
    for (int j = 0; j < BN; ++j) {
      acc[0][j] = __builtin_amdgcn_wmma_f32_16x16x32_bf16(false, a0c, false, bcur[j], (short)0, acc[0][j], false, false);
      acc[1][j] = __builtin_amdgcn_wmma_f32_16x16x32_bf16(false, a1c, false, bcur[j], (short)0, acc[1][j], false, false);
    }
    // prefetch next trip's stage A (k0+64) straight into stage-A buffers
    // (dead over-read on the final trip; buffers carry 256B tail slack)
    a0c = frag_ld(pa0 + 64);
    a1c = frag_ld(pa1 + 64);
#pragma unroll
    for (int j = 0; j < BN; ++j) bcur[j] = frag_ld(pb[j] + 64);
    // compute stage B
#pragma unroll
    for (int j = 0; j < BN; ++j) {
      acc[0][j] = __builtin_amdgcn_wmma_f32_16x16x32_bf16(false, a0n, false, bnx[j], (short)0, acc[0][j], false, false);
      acc[1][j] = __builtin_amdgcn_wmma_f32_16x16x32_bf16(false, a1n, false, bnx[j], (short)0, acc[1][j], false, false);
    }
    pa0 += 64; pa1 += 64;
#pragma unroll
    for (int j = 0; j < BN; ++j) pb[j] += 64;
  }

  const int rowAdd = (lane >> 4) ? 8 : 0;
  const int colIn  = lane & 15;
#pragma unroll
  for (int mi = 0; mi < 2; ++mi) {
#pragma unroll
    for (int j = 0; j < BN; ++j) {
      const int cb = n0 + j * 16 + colIn;
      if (cb >= N) continue;
      const float bv = biasp ? biasp[cb] : 0.0f;
      for (int rr = 0; rr < 8; ++rr) {
        const int m = m0 + mi * 16 + rowAdd + rr;
        if (m >= M) continue;
        float v = acc[mi][j][rr] + bv;
        size_t idx;
        if (CT) {
          const int bb = m / S_;                 // batch index
          const int ss = m - bb * S_;            // token index
          idx = coff + (size_t)bb * ((size_t)NH_ * DH_ * SP_K)
                     + (size_t)cb * SP_K + (size_t)ss;
        } else {
          const size_t crow = CMODE ? (size_t)(m + m / 196 + 1) : (size_t)m;
          idx = coff + crow * (size_t)ldc + (size_t)cb;
        }
        if (RESID)    v += resid[idx];
        if (GELU_ACT) v = 0.5f * v * (1.0f + erff(v * 0.70710678118f));
        if (OUT_BF16) ((__bf16*)Cv)[idx] = (__bf16)v;
        else          ((float*)Cv)[idx]  = v;
      }
    }
  }
}

// ---------------------------------------------------------------------------
// Elementwise / reduction kernels
// ---------------------------------------------------------------------------
__global__ void f32_to_bf16_kernel(const float* __restrict__ in, __bf16* __restrict__ out, size_t n) {
  size_t i = (size_t)blockIdx.x * blockDim.x + threadIdx.x;
  if (i < n) out[i] = (__bf16)in[i];
}

__global__ void patchify_kernel(const float* __restrict__ img, __bf16* __restrict__ out, int total) {
  int i = blockIdx.x * blockDim.x + threadIdx.x;
  if (i >= total) return;
  int kk = i % D_;
  int t  = i / D_;
  int b  = t / NPATCH;
  int p  = t - b * NPATCH;
  int py = p / P_, px = p - py * P_;
  int c   = kk >> 8;          // / 256
  int rem = kk & 255;
  int iy = rem >> 4, ix = rem & 15;
  int y = py * 16 + iy, x = px * 16 + ix;
  out[i] = (__bf16)img[(((size_t)b * 3 + c) * 224 + y) * 224 + x];
}

__global__ void addpos_kernel(float* __restrict__ x, const float* __restrict__ cls,
                              const float* __restrict__ pos, int total) {
  int i = blockIdx.x * blockDim.x + threadIdx.x;
  if (i >= total) return;
  int d = i % D_;
  int r = i / D_;
  int s = r % S_;
  if (s == 0) x[i] = cls[d] + pos[d];
  else        x[i] += pos[(size_t)s * D_ + d];
}

__global__ void layernorm_kernel(const float* __restrict__ x, const float* __restrict__ g,
                                 const float* __restrict__ bta, __bf16* __restrict__ out) {
  __shared__ float red[256];
  const int row = blockIdx.x;
  const float* xr = x + (size_t)row * D_;
  float s = 0.f;
  for (int j = threadIdx.x; j < D_; j += 256) s += xr[j];
  red[threadIdx.x] = s; __syncthreads();
  for (int st = 128; st > 0; st >>= 1) {
    if (threadIdx.x < st) red[threadIdx.x] += red[threadIdx.x + st];
    __syncthreads();
  }
  const float mu = red[0] * (1.0f / D_);
  __syncthreads();
  float v = 0.f;
  for (int j = threadIdx.x; j < D_; j += 256) { float d = xr[j] - mu; v += d * d; }
  red[threadIdx.x] = v; __syncthreads();
  for (int st = 128; st > 0; st >>= 1) {
    if (threadIdx.x < st) red[threadIdx.x] += red[threadIdx.x + st];
    __syncthreads();
  }
  const float rstd = rsqrtf(red[0] * (1.0f / D_) + 1e-5f);
  __bf16* orow = out + (size_t)row * D_;
  for (int j = threadIdx.x; j < D_; j += 256)
    orow[j] = (__bf16)((xr[j] - mu) * rstd * g[j] + bta[j]);
}

// rows grid.x = SP_M per (b,h); writes zero-padded bf16 attn rows of width SP_K
__global__ void softmax_kernel(const float* __restrict__ scores, __bf16* __restrict__ attn) {
  __shared__ float red[256];
  const int r = blockIdx.x;
  const int tid = threadIdx.x;
  const size_t base = ((size_t)blockIdx.y * SP_M + r) * SP_K;
  if (r >= S_) { if (tid < SP_K) attn[base + tid] = (__bf16)0.0f; return; }
  const float scale = 1.0f / (float)(DH_ * DH_);   // reference scales by d_head**2
  float v = -3.4e38f;
  if (tid < S_) v = scores[base + tid] * scale;
  red[tid] = v; __syncthreads();
  for (int st = 128; st > 0; st >>= 1) {
    if (tid < st) red[tid] = fmaxf(red[tid], red[tid + st]);
    __syncthreads();
  }
  const float mx = red[0]; __syncthreads();
  float e = 0.f;
  if (tid < S_) e = __expf(v - mx);
  red[tid] = e; __syncthreads();
  for (int st = 128; st > 0; st >>= 1) {
    if (tid < st) red[tid] += red[tid + st];
    __syncthreads();
  }
  const float inv = 1.0f / red[0];
  if (tid < SP_K) attn[base + tid] = (__bf16)((tid < S_) ? e * inv : 0.0f);
}

// ---------------------------------------------------------------------------
// Host driver
// ---------------------------------------------------------------------------
extern "C" void kernel_launch(void* const* d_in, const int* in_sizes, int n_in,
                              void* d_out, int out_size, void* d_ws, size_t ws_size,
                              hipStream_t stream) {
  (void)in_sizes; (void)n_in; (void)out_size; (void)ws_size;
  const float* images  = (const float*)d_in[0];
  const float* mapperw = (const float*)d_in[1];
  const float* mapperb = (const float*)d_in[2];
  const float* clstok  = (const float*)d_in[3];
  const float* posemb  = (const float*)d_in[4];
  const float* ln1g = (const float*)d_in[5];
  const float* ln1b = (const float*)d_in[6];
  const float* qw = (const float*)d_in[7];
  const float* qb = (const float*)d_in[8];
  const float* kw = (const float*)d_in[9];
  const float* kb = (const float*)d_in[10];
  const float* vw = (const float*)d_in[11];
  const float* vb = (const float*)d_in[12];
  const float* ow = (const float*)d_in[13];
  const float* ob = (const float*)d_in[14];
  const float* ln2g = (const float*)d_in[15];
  const float* ln2b = (const float*)d_in[16];
  const float* w1 = (const float*)d_in[17];
  const float* b1 = (const float*)d_in[18];
  const float* w2 = (const float*)d_in[19];
  const float* b2 = (const float*)d_in[20];

  const int Mtok = B_ * NPATCH;   // 12544
  const int Mx   = B_ * S_;       // 12608 (multiple of 16)

  char* ws = (char*)d_ws;
  size_t off = 0;
  // every allocation gets 256B tail slack for the GEMM ping-pong over-read
  auto alloc = [&](size_t bytes) { size_t o = off; off += (bytes + 256 + 255) & ~(size_t)255; return o; };

  __bf16* wb_map = (__bf16*)(ws + alloc((size_t)D_ * D_ * 2));
  __bf16* wb_q   = (__bf16*)(ws + alloc((size_t)L_ * NH_ * DH_ * DH_ * 2));
  __bf16* wb_k   = (__bf16*)(ws + alloc((size_t)L_ * NH_ * DH_ * DH_ * 2));
  __bf16* wb_v   = (__bf16*)(ws + alloc((size_t)L_ * NH_ * DH_ * DH_ * 2));
  __bf16* wb_o   = (__bf16*)(ws + alloc((size_t)L_ * D_ * D_ * 2));
  __bf16* wb_1   = (__bf16*)(ws + alloc((size_t)L_ * FF_ * D_ * 2));
  __bf16* wb_2   = (__bf16*)(ws + alloc((size_t)L_ * D_ * FF_ * 2));
  __bf16* patches= (__bf16*)(ws + alloc((size_t)Mtok * D_ * 2));
  float*  x      = (float*) (ws + alloc((size_t)Mx * D_ * 4));
  __bf16* xln    = (__bf16*)(ws + alloc((size_t)Mx * D_ * 2));
  __bf16* qbuf   = (__bf16*)(ws + alloc((size_t)(Mx + 64) * D_ * 2));  // +pad rows: padded-N reads
  __bf16* kbuf   = (__bf16*)(ws + alloc((size_t)(Mx + 64) * D_ * 2));
  __bf16* vt     = (__bf16*)(ws + alloc((size_t)B_ * NH_ * DH_ * SP_K * 2)); // V^T [b,h,e,s_pad]
  __bf16* ctx    = (__bf16*)(ws + alloc((size_t)Mx * D_ * 2));
  size_t scores_off = alloc((size_t)B_ * NH_ * SP_M * SP_K * 4);       // 164 MB
  float*  scores = (float*)(ws + scores_off);
  __bf16* mid    = (__bf16*)(ws + scores_off);                          // alias (77 MB, disjoint in time)
  __bf16* attn   = (__bf16*)(ws + alloc((size_t)B_ * NH_ * SP_M * SP_K * 2));

  auto cvt = [&](const float* src, __bf16* dst, size_t n) {
    f32_to_bf16_kernel<<<(unsigned)((n + 255) / 256), 256, 0, stream>>>(src, dst, n);
  };
  cvt(mapperw, wb_map, (size_t)D_ * D_);
  cvt(qw, wb_q, (size_t)L_ * NH_ * DH_ * DH_);
  cvt(kw, wb_k, (size_t)L_ * NH_ * DH_ * DH_);
  cvt(vw, wb_v, (size_t)L_ * NH_ * DH_ * DH_);
  cvt(ow, wb_o, (size_t)L_ * D_ * D_);
  cvt(w1, wb_1, (size_t)L_ * FF_ * D_);
  cvt(w2, wb_2, (size_t)L_ * D_ * FF_);

  // patchify -> bf16 [B*196, 768]
  {
    int total = Mtok * D_;
    patchify_kernel<<<(total + 255) / 256, 256, 0, stream>>>(images, patches, total);
  }

  const dim3 blk(256);
  // mapper GEMM: tokens = patches @ mapper_w.T + b, row-remapped into x (skip cls rows)
  gemm_wmma_kernel<4, 0, 0, 0, 1, 0><<<dim3(D_ / 128, Mtok / 128, 1), blk, 0, stream>>>(
      patches, D_, 0, 0, wb_map, D_, 0, 0, x, D_, 0, 0,
      mapperb, 0, 0, nullptr, Mtok, D_, D_, 1);

  // cls row + positional embedding
  {
    int total = Mx * D_;
    addpos_kernel<<<(total + 255) / 256, 256, 0, stream>>>(x, clstok, posemb, total);
  }

  const int gy = (Mx + 127) / 128;   // 99
  int eidx = 0;
  for (int i = 0; i < L_; ++i) {
    // --- LN1 ---
    layernorm_kernel<<<Mx, 256, 0, stream>>>(x, ln1g + (size_t)i * D_, ln1b + (size_t)i * D_, xln);

    // --- per-head QKV (batched over heads; K=N=64) ---
    const size_t wqoff = (size_t)i * NH_ * DH_ * DH_;
    gemm_wmma_kernel<2, 1, 0, 0, 0, 0><<<dim3(1, gy, NH_), blk, 0, stream>>>(
        xln, D_, 0, DH_, wb_q + wqoff, DH_, 0, (long long)DH_ * DH_,
        qbuf, D_, 0, DH_, qb + (size_t)i * NH_ * DH_, 0, DH_, nullptr, Mx, DH_, DH_, NH_);
    gemm_wmma_kernel<2, 1, 0, 0, 0, 0><<<dim3(1, gy, NH_), blk, 0, stream>>>(
        xln, D_, 0, DH_, wb_k + wqoff, DH_, 0, (long long)DH_ * DH_,
        kbuf, D_, 0, DH_, kb + (size_t)i * NH_ * DH_, 0, DH_, nullptr, Mx, DH_, DH_, NH_);
    // V with transposed store into vt[b,h,e,s] (s padded to SP_K)
    gemm_wmma_kernel<2, 1, 0, 0, 0, 1><<<dim3(1, gy, NH_), blk, 0, stream>>>(
        xln, D_, 0, DH_, wb_v + wqoff, DH_, 0, (long long)DH_ * DH_,
        vt, SP_K, 0, (long long)DH_ * SP_K,
        vb + (size_t)i * NH_ * DH_, 0, DH_, nullptr, Mx, DH_, DH_, NH_);

    // --- scores = Q @ K.T   (batched over B*NH; M=S, N=208, K=64) ---
    gemm_wmma_kernel<4, 0, 0, 0, 0, 0><<<dim3((SP_M + 127) / 128, (S_ + 127) / 128, B_ * NH_), blk, 0, stream>>>(
        qbuf, D_, (long long)S_ * D_, DH_,
        kbuf, D_, (long long)S_ * D_, DH_,
        scores, SP_K, (long long)NH_ * SP_M * SP_K, (long long)SP_M * SP_K,
        nullptr, 0, 0, nullptr, S_, SP_M, DH_, NH_);

    // --- softmax (scale 1/DH^2, zero-pad to SP_K) ---
    softmax_kernel<<<dim3(SP_M, B_ * NH_), 256, 0, stream>>>(scores, attn);

    // --- ctx = attn @ vt.T  (K-contiguous both sides; K=256 zero-padded) ---
    gemm_wmma_kernel<2, 1, 0, 0, 0, 0><<<dim3(1, (S_ + 127) / 128, B_ * NH_), blk, 0, stream>>>(
        attn, SP_K, (long long)NH_ * SP_M * SP_K, (long long)SP_M * SP_K,
        vt,   SP_K, (long long)NH_ * DH_ * SP_K,  (long long)DH_ * SP_K,
        ctx, D_, (long long)S_ * D_, DH_,
        nullptr, 0, 0, nullptr, S_, DH_, SP_K, NH_);

    // --- out projection + bias + residual into x ---
    gemm_wmma_kernel<4, 0, 0, 1, 0, 0><<<dim3(D_ / 128, gy, 1), blk, 0, stream>>>(
        ctx, D_, 0, 0, wb_o + (size_t)i * D_ * D_, D_, 0, 0,
        x, D_, 0, 0, ob + (size_t)i * D_, 0, 0, x, Mx, D_, D_, 1);

    // --- LN2 ---
    layernorm_kernel<<<Mx, 256, 0, stream>>>(x, ln2g + (size_t)i * D_, ln2b + (size_t)i * D_, xln);

    // --- MLP1: mid = gelu(h2 @ w1.T + b1), bf16 out ---
    gemm_wmma_kernel<4, 1, 1, 0, 0, 0><<<dim3(FF_ / 128, gy, 1), blk, 0, stream>>>(
        xln, D_, 0, 0, wb_1 + (size_t)i * FF_ * D_, D_, 0, 0,
        mid, FF_, 0, 0, b1 + (size_t)i * FF_, 0, 0, nullptr, Mx, FF_, D_, 1);

    // --- MLP2: x += mid @ w2.T + b2 ---
    gemm_wmma_kernel<4, 0, 0, 1, 0, 0><<<dim3(D_ / 128, gy, 1), blk, 0, stream>>>(
        mid, FF_, 0, 0, wb_2 + (size_t)i * D_ * FF_, FF_, 0, 0,
        x, D_, 0, 0, b2 + (size_t)i * D_, 0, 0, x, Mx, D_, FF_, 1);

    // --- extract ---
    if (i == 2 || i == 5 || i == 8 || i == 11) {
      hipMemcpyAsync((float*)d_out + (size_t)eidx * Mx * D_, x,
                     (size_t)Mx * D_ * sizeof(float), hipMemcpyDeviceToDevice, stream);
      ++eidx;
    }
  }
}
